// MultiSwinTrans_24017457119676
// MI455X (gfx1250) — matrix operand order repaired and checked by reading, not compiled, not run on hardware
//
#include <hip/hip_runtime.h>
#include <math.h>

// ---------------------------------------------------------------------------
// CDNA5 (gfx1250) wave32 WMMA implementation of the SwinViT reference.
// GEMMs: v_wmma_f32_16x16x32_f16 (f16 in, f32 acc).
// Generic GEMM stages A/B tiles via the Tensor Data Mover
// (tensor_load_to_lds + s_wait_tensorcnt), descriptors per ISA 08 §8.
// ---------------------------------------------------------------------------

typedef __attribute__((ext_vector_type(16))) _Float16     v16h;
typedef __attribute__((ext_vector_type(8)))  float        v8f;
typedef __attribute__((ext_vector_type(4)))  unsigned int v4u;
typedef __attribute__((ext_vector_type(8)))  int          v8i;
typedef __attribute__((ext_vector_type(4)))  int          v4i;

// ---- WMMA fragment loaders (contiguous k = 16h + s order, same for A and B) -----
// A 16x32 f16 row-major in LDS; lane L: m=L&15, h=L>>4,
// VGPR v holds K = h*16+2v, K+1
__device__ inline v16h load_A_frag(const _Float16* sA) {
  int lane = threadIdx.x & 31;
  int m = lane & 15, h = lane >> 4;
  unsigned int tmp[8];
#pragma unroll
  for (int v = 0; v < 8; ++v) {
    int k = h * 16 + 2 * v;
    tmp[v] = *(const unsigned int*)(sA + m * 32 + k);
  }
  v16h out; __builtin_memcpy(&out, tmp, 32);
  return out;
}

// B 32x16 f16 (K rows x N cols) row-major in LDS; lane: n=L&15, h=L>>4,
// VGPR v holds K = h*16+2v, +1 at column n
__device__ inline v16h load_B_frag(const _Float16* sB) {
  int lane = threadIdx.x & 31;
  int n = lane & 15, h = lane >> 4;
  _Float16 tmp[16];
#pragma unroll
  for (int v = 0; v < 8; ++v) {
    int k = h * 16 + 2 * v;
    tmp[2 * v]     = sB[k * 16 + n];
    tmp[2 * v + 1] = sB[(k + 1) * 16 + n];
  }
  v16h out; __builtin_memcpy(&out, tmp, 32);
  return out;
}

// A fragment from a raw fp32 16(m) x 32(k) TDM-loaded tile (cvt fused)
__device__ inline v16h load_A_frag_f32(const float* sA) {
  int lane = threadIdx.x & 31;
  int m = lane & 15, h = lane >> 4;
  _Float16 tmp[16];
#pragma unroll
  for (int v = 0; v < 8; ++v) {
    int k = h * 16 + 2 * v;
    tmp[2 * v]     = (_Float16)sA[m * 32 + k];
    tmp[2 * v + 1] = (_Float16)sA[m * 32 + k + 1];
  }
  v16h out; __builtin_memcpy(&out, tmp, 32);
  return out;
}

// B fragment from a raw fp32 16(n) x 32(k) TDM-loaded weight tile:
// B[k][n] = tile[n][k] (transpose read fused with cvt)
__device__ inline v16h load_Bnk_frag_f32(const float* sB) {
  int lane = threadIdx.x & 31;
  int n = lane & 15, h = lane >> 4;
  _Float16 tmp[16];
#pragma unroll
  for (int v = 0; v < 8; ++v) {
    int k = h * 16 + 2 * v;
    tmp[2 * v]     = (_Float16)sB[n * 32 + k];
    tmp[2 * v + 1] = (_Float16)sB[n * 32 + k + 1];
  }
  v16h out; __builtin_memcpy(&out, tmp, 32);
  return out;
}
// C/D: lane L -> col n = L&15 ; VGPR v -> row m = (L>>4)*8 + v

// ---------------------------------------------------------------------------
// Tensor Data Mover: load a 2D fp32 tile (tile_d0 x tile_d1, row stride
// stride_elems) from global into LDS (packed, row-major, d0-contiguous).
// tensor_d0/d1 = remaining extent from tile origin -> TDM zero-fills OOB.
// D# layout per CDNA5 ISA. Groups 2/3 unused (2D).
// ---------------------------------------------------------------------------
__device__ inline void tdm_load_tile_f32(unsigned lds_off, const float* gptr,
                                         unsigned tensor_d0, unsigned tensor_d1,
                                         unsigned tile_d0, unsigned tile_d1,
                                         unsigned long long stride_elems) {
  unsigned long long ga = (unsigned long long)(const void*)gptr;
  v4u g0;
  g0[0] = 1u;                                            // count=1, user mode
  g0[1] = lds_off;                                       // lds_addr (bytes)
  g0[2] = (unsigned)(ga & 0xffffffffull);                // global_addr[31:0]
  g0[3] = (unsigned)((ga >> 32) & 0x1ffffffull) | (2u << 30);  // [56:32] | type=2
  v8i g1;
  g1[0] = (int)(2u << 16);                               // data_size=4B, mask=0
  g1[1] = (int)((tensor_d0 & 0xffffu) << 16);            // tensor_dim0[15:0]
  g1[2] = (int)(((tensor_d0 >> 16) & 0xffffu) | ((tensor_d1 & 0xffffu) << 16));
  g1[3] = (int)(((tensor_d1 >> 16) & 0xffffu) | (tile_d0 << 16));  // tile_dim0
  g1[4] = (int)(tile_d1 & 0xffffu);                      // tile_dim1 (tile_dim2=0)
  g1[5] = (int)(stride_elems & 0xffffffffull);           // dim0_stride[31:0]
  g1[6] = (int)((stride_elems >> 32) & 0xffffull);       // dim0_stride[47:32]
  g1[7] = 0;
  v4i z4 = {};
  v8i z8 = {};
  __builtin_amdgcn_tensor_load_to_lds(g0, g1, z4, z4, z8, 0);
}

// ---------------------------------------------------------------------------
// Generic GEMM: Out[M x N] (row stride ldo) = act( X[M x K] @ W^T + bias )
// W stored (N x K) row-major. TDM-staged tiles.  ACT: 0=none 1=gelu(exact)
// ---------------------------------------------------------------------------
template <int ACT>
__global__ void gemm_xwt_kernel(const float* __restrict__ X, const float* __restrict__ W,
                                const float* __restrict__ bias, float* __restrict__ Out,
                                int M, int N, int K, int ldo) {
  __shared__ float sA32[16 * 32];
  __shared__ float sB32[16 * 32];
  int mt = blockIdx.x, nt = blockIdx.y;
  int lane = threadIdx.x;
  unsigned ldsA = (unsigned)(unsigned long long)(void*)sA32;
  unsigned ldsB = (unsigned)(unsigned long long)(void*)sB32;
  v8f acc = {};
  for (int k0 = 0; k0 < K; k0 += 32) {
    if (lane == 0) {
      tdm_load_tile_f32(ldsA, X + (size_t)mt * 16 * K + k0,
                        (unsigned)(K - k0), (unsigned)(M - mt * 16), 32u, 16u,
                        (unsigned long long)K);
      tdm_load_tile_f32(ldsB, W + (size_t)nt * 16 * K + k0,
                        (unsigned)(K - k0), (unsigned)(N - nt * 16), 32u, 16u,
                        (unsigned long long)K);
    }
    __builtin_amdgcn_s_wait_tensorcnt(0);
    __syncthreads();
    acc = __builtin_amdgcn_wmma_f32_16x16x32_f16(false, load_A_frag_f32(sA32), false,
                                                 load_Bnk_frag_f32(sB32), (short)0, acc,
                                                 false, false);
    __syncthreads();
  }
  int n = (lane & 15) + nt * 16;
  int h = lane >> 4;
  if (n < N) {
    float bv = bias ? bias[n] : 0.f;
#pragma unroll
    for (int v = 0; v < 8; ++v) {
      int m = mt * 16 + h * 8 + v;
      if (m < M) {
        float r = acc[v] + bv;
        if (ACT == 1) r = 0.5f * r * (1.f + erff(r * 0.70710678118f));
        Out[(size_t)m * ldo + n] = r;
      }
    }
  }
}

// ---------------------------------------------------------------------------
// Patch embed: Conv3d(1->48, k=16, s=16) as im2col GEMM.  M=8192 N=48 K=4096.
// Gather pattern is 3D-strided per token -> manual staging, batched loads.
// ---------------------------------------------------------------------------
__global__ void patch_embed_kernel(const float* __restrict__ x, const float* __restrict__ W,
                                   const float* __restrict__ bias, float* __restrict__ Out) {
  const int N = 48, K = 4096;
  __shared__ _Float16 sA[512];
  __shared__ _Float16 sB[512];
  int mt = blockIdx.x, nt = blockIdx.y;
  int lane = threadIdx.x;
  v8f acc = {};
  for (int k0 = 0; k0 < K; k0 += 32) {
    // load phase (16 independent loads/lane), then store phase
    float av[16];
#pragma unroll
    for (int i = 0; i < 16; ++i) {
      int t = mt * 16 + i;          // row = i, col = lane
      int k = k0 + lane;
      int b = t >> 12, tz = (t >> 8) & 15, ty = (t >> 4) & 15, tx = t & 15;
      int ii = k >> 8, j = (k >> 4) & 15, kk = k & 15;
      size_t gi = (((size_t)b * 256 + (tz * 16 + ii)) * 256 + (ty * 16 + j)) * 256 + (tx * 16 + kk);
      av[i] = x[gi];
    }
    float bv[16];
#pragma unroll
    for (int i = 0; i < 16; ++i) {
      int e = i * 32 + lane;
      int kk2 = e >> 4, nn = e & 15;
      int gn = nt * 16 + nn;
      bv[i] = (gn < N) ? W[(size_t)gn * K + k0 + kk2] : 0.f;
    }
#pragma unroll
    for (int i = 0; i < 16; ++i) sA[i * 32 + lane] = (_Float16)av[i];
#pragma unroll
    for (int i = 0; i < 16; ++i) sB[i * 32 + lane] = (_Float16)bv[i];
    __syncthreads();
    acc = __builtin_amdgcn_wmma_f32_16x16x32_f16(false, load_A_frag(sA), false,
                                                 load_B_frag(sB), (short)0, acc, false, false);
    __syncthreads();
  }
  int n = (lane & 15) + nt * 16, h = lane >> 4;
  if (n < N) {
#pragma unroll
    for (int v = 0; v < 8; ++v) {
      int m = mt * 16 + h * 8 + v;
      Out[(size_t)m * N + n] = acc[v] + bias[n];
    }
  }
}

// ---------------------------------------------------------------------------
// LayerNorm over last dim C (one wave per row).  g==nullptr -> plain LN.
// ---------------------------------------------------------------------------
__global__ void layernorm_kernel(const float* __restrict__ X, const float* __restrict__ g,
                                 const float* __restrict__ b, float* __restrict__ Out,
                                 int C) {
  int row = blockIdx.x;
  int lane = threadIdx.x;
  const float* xr = X + (size_t)row * C;
  float s = 0.f, s2 = 0.f;
  for (int i = lane; i < C; i += 32) { float v = xr[i]; s += v; s2 += v * v; }
#pragma unroll
  for (int o = 16; o > 0; o >>= 1) { s += __shfl_xor(s, o, 32); s2 += __shfl_xor(s2, o, 32); }
  float m = s / C;
  float rstd = rsqrtf(s2 / C - m * m + 1e-5f);
  float* orow = Out + (size_t)row * C;
  for (int i = lane; i < C; i += 32) {
    float v = (xr[i] - m) * rstd;
    if (g) v = v * g[i] + b[i];
    orow[i] = v;
  }
}

// ---------------------------------------------------------------------------
// Window partition fused with zero-pad + cyclic shift (roll by -ss).
// ---------------------------------------------------------------------------
__global__ void win_gather_kernel(const float* __restrict__ Xn, float* __restrict__ XW,
                                  int D, int C, int ws, int ss, int pd) {
  int nwd = pd / ws;
  int n = ws * ws * ws;
  int wb = blockIdx.x;
  int tok = blockIdx.y;
  int lane = threadIdx.x;
  int nw = nwd * nwd * nwd;
  int b = wb / nw, w = wb % nw;
  int wz = w / (nwd * nwd), wy = (w / nwd) % nwd, wx = w % nwd;
  int iz = tok / (ws * ws), iy = (tok / ws) % ws, ix = tok % ws;
  int sz = (wz * ws + iz + ss) % pd, sy = (wy * ws + iy + ss) % pd, sx = (wx * ws + ix + ss) % pd;
  bool in = (sz < D) && (sy < D) && (sx < D);
  float* out = XW + ((size_t)wb * n + tok) * C;
  const float* src = Xn + ((size_t)((b * D + sz) * D + sy) * D + sx) * C;
  for (int c = lane; c < C; c += 32) out[c] = in ? src[c] : 0.f;
}

// Window reverse + un-shift + crop + residual add (in place on X).
__global__ void win_scatter_kernel(const float* __restrict__ AW, float* __restrict__ X,
                                   int D, int C, int ws, int ss, int pd) {
  int t = blockIdx.x;
  int lane = threadIdx.x;
  int b = t / (D * D * D), rr = t % (D * D * D);
  int z = rr / (D * D), y = (rr / D) % D, x = rr % D;
  int nwd = pd / ws, nw = nwd * nwd * nwd, n = ws * ws * ws;
  int qz = (z - ss + pd) % pd, qy = (y - ss + pd) % pd, qx = (x - ss + pd) % pd;
  int w = ((qz / ws) * nwd + (qy / ws)) * nwd + (qx / ws);
  int tok = ((qz % ws) * ws + (qy % ws)) * ws + (qx % ws);
  const float* src = AW + ((size_t)(b * nw + w) * n + tok) * C;
  float* dst = X + (size_t)t * C;
  for (int c = lane; c < C; c += 32) dst[c] += src[c];
}

// ---------------------------------------------------------------------------
// Attention scores: attn = (q*0.25) . k^T + rpb (+ shift mask). head_dim=16.
// ---------------------------------------------------------------------------
__device__ inline int reg3(int p, int pd, int ws, int ss) {
  return (p < pd - ws) ? 0 : ((p < pd - ss) ? 1 : 2);
}

__global__ void attn_score_kernel(const float* __restrict__ QKV, const float* __restrict__ rpb,
                                  float* __restrict__ ATT, int n, int C, int heads,
                                  int nwd, int ws, int ss, int pd, int shifted) {
  int mt = blockIdx.x, ntile = blockIdx.y;
  int wh = blockIdx.z;
  int h = wh % heads, wb = wh / heads;
  int lane = threadIdx.x;
  __shared__ _Float16 sA[512];
  __shared__ _Float16 sB[512];
  const float scale = 0.25f;
  const size_t rs = 3 * (size_t)C;
  const float* base = QKV + (size_t)wb * n * 3 * C;
  // A: row i = token mt*16+i, col lane (K=16 valid of 32)
  float av[16];
#pragma unroll
  for (int i = 0; i < 16; ++i) {
    int tm = mt * 16 + i;
    av[i] = (lane < 16 && tm < n) ? base[tm * rs + h * 16 + lane] * scale : 0.f;
  }
  float bv[16];
#pragma unroll
  for (int i = 0; i < 16; ++i) {
    int e = i * 32 + lane;
    int kk = e >> 4, nn = e & 15;
    int tn = ntile * 16 + nn;
    bv[i] = (kk < 16 && tn < n) ? base[tn * rs + C + h * 16 + kk] : 0.f;
  }
#pragma unroll
  for (int i = 0; i < 16; ++i) sA[i * 32 + lane] = (_Float16)av[i];
#pragma unroll
  for (int i = 0; i < 16; ++i) sB[i * 32 + lane] = (_Float16)bv[i];
  __syncthreads();
  v8f acc = {};
  acc = __builtin_amdgcn_wmma_f32_16x16x32_f16(false, load_A_frag(sA), false,
                                               load_B_frag(sB), (short)0, acc, false, false);
  int nn = (lane & 15) + ntile * 16;
  int hh = lane >> 4;
  if (nn < n) {
    int nz7 = nn / 49, ny7 = (nn / 7) % 7, nx7 = nn % 7;
    int nw_ = nwd * nwd * nwd;
    int w = wb % nw_;
    int wz = w / (nwd * nwd), wy = (w / nwd) % nwd, wx = w % nwd;
    int rnz = 0, rny = 0, rnx = 0;
    if (shifted) {
      rnz = reg3(wz * ws + nn / (ws * ws), pd, ws, ss);
      rny = reg3(wy * ws + (nn / ws) % ws, pd, ws, ss);
      rnx = reg3(wx * ws + nn % ws, pd, ws, ss);
    }
    float* orow = ATT + ((size_t)wb * heads + h) * n * n;
#pragma unroll
    for (int v = 0; v < 8; ++v) {
      int m = mt * 16 + hh * 8 + v;
      if (m < n) {
        int mz7 = m / 49, my7 = (m / 7) % 7, mx7 = m % 7;
        int idx = ((mz7 - nz7 + 6) * 13 + (my7 - ny7 + 6)) * 13 + (mx7 - nx7 + 6);
        float val = acc[v] + rpb[idx * heads + h];
        if (shifted) {
          int rmz = reg3(wz * ws + m / (ws * ws), pd, ws, ss);
          int rmy = reg3(wy * ws + (m / ws) % ws, pd, ws, ss);
          int rmx = reg3(wx * ws + m % ws, pd, ws, ss);
          if (!(rmz == rnz && rmy == rny && rmx == rnx)) val -= 100.f;
        }
        orow[(size_t)m * n + nn] = val;
      }
    }
  }
}

// Row softmax (one wave per row of length n).
__global__ void softmax_kernel(float* __restrict__ ATT, int n) {
  float* r = ATT + (size_t)blockIdx.x * n;
  int lane = threadIdx.x;
  float mx = -3.4e38f;
  for (int i = lane; i < n; i += 32) mx = fmaxf(mx, r[i]);
#pragma unroll
  for (int o = 16; o > 0; o >>= 1) mx = fmaxf(mx, __shfl_xor(mx, o, 32));
  float s = 0.f;
  for (int i = lane; i < n; i += 32) { float e = __expf(r[i] - mx); r[i] = e; s += e; }
#pragma unroll
  for (int o = 16; o > 0; o >>= 1) s += __shfl_xor(s, o, 32);
  float inv = 1.f / s;
  for (int i = lane; i < n; i += 32) r[i] *= inv;
}

// out = attn @ v ; N = head_dim = 16, K loop over n (zero-padded)
__global__ void attn_av_kernel(const float* __restrict__ ATT, const float* __restrict__ QKV,
                               float* __restrict__ AV, int n, int C, int heads) {
  int mt = blockIdx.x;
  int wh = blockIdx.y;
  int h = wh % heads, wb = wh / heads;
  int lane = threadIdx.x;
  __shared__ _Float16 sA[512];
  __shared__ _Float16 sB[512];
  const float* arow = ATT + ((size_t)wb * heads + h) * n * n;
  const float* vbase = QKV + (size_t)wb * n * 3 * C + 2 * C + h * 16;
  v8f acc = {};
  for (int k0 = 0; k0 < n; k0 += 32) {
    float av[16];
#pragma unroll
    for (int i = 0; i < 16; ++i) {
      int m = mt * 16 + i, kk = k0 + lane;
      av[i] = (m < n && kk < n) ? arow[(size_t)m * n + kk] : 0.f;
    }
    float bv[16];
#pragma unroll
    for (int i = 0; i < 16; ++i) {
      int e = i * 32 + lane;
      int kk = e >> 4, nn = e & 15;
      int tok = k0 + kk;
      bv[i] = (tok < n) ? vbase[(size_t)tok * 3 * C + nn] : 0.f;
    }
#pragma unroll
    for (int i = 0; i < 16; ++i) sA[i * 32 + lane] = (_Float16)av[i];
#pragma unroll
    for (int i = 0; i < 16; ++i) sB[i * 32 + lane] = (_Float16)bv[i];
    __syncthreads();
    acc = __builtin_amdgcn_wmma_f32_16x16x32_f16(false, load_A_frag(sA), false,
                                                 load_B_frag(sB), (short)0, acc, false, false);
    __syncthreads();
  }
  int nn = lane & 15, hh = lane >> 4;
#pragma unroll
  for (int v = 0; v < 8; ++v) {
    int m = mt * 16 + hh * 8 + v;
    if (m < n) AV[((size_t)wb * n + m) * C + h * 16 + nn] = acc[v];
  }
}

// Elementwise residual add: X += Y
__global__ void add_kernel(float* __restrict__ X, const float* __restrict__ Y, size_t N) {
  size_t i = (size_t)blockIdx.x * blockDim.x + threadIdx.x;
  if (i < N) X[i] += Y[i];
}

// Patch-merge gather: (b,D,D,D,C) -> (b,D/2,D/2,D/2,8C), part order i*4+j*2+k
__global__ void merge_gather_kernel(const float* __restrict__ X, float* __restrict__ M8,
                                    int D, int C) {
  int Dh = D / 2;
  int t = blockIdx.x;
  int lane = threadIdx.x;
  int b = t / (Dh * Dh * Dh), rr = t % (Dh * Dh * Dh);
  int z = rr / (Dh * Dh), y = (rr / Dh) % Dh, x = rr % Dh;
  float* out = M8 + (size_t)t * 8 * C;
  for (int e = lane; e < 8 * C; e += 32) {
    int p = e / C, ch = e % C;
    int i = (p >> 2) & 1, j = (p >> 1) & 1, k = p & 1;
    out[e] = X[((size_t)((b * D + 2 * z + i) * D + (2 * y + j)) * D + (2 * x + k)) * C + ch];
  }
}

// ---------------------------------------------------------------------------
// Host orchestration (jax pytree flatten order: dict keys sorted, lists in order)
//  0:x 1:c 2:clin_b 3:clin_w 4:conv_b 5:conv_w
//  6..17: fuse{c1_b,c1_w,c2_b,c2_w,s0a_b,s0a_w,s0b_b,s0b_w,s1a_b,s1a_w,s1b_b,s1b_w}
//  18..121: layers[l][blk]{fc1_b,fc1_w,fc2_b,fc2_w,n1_b,n1_g,n2_b,n2_g,proj_b,proj_w,qkv_b,qkv_w,rpb}
//  122..133: merges[l]{n_b,n_g,red_w}
// ---------------------------------------------------------------------------
#define PW(i) ((const float*)d_in[(i)])

extern "C" void kernel_launch(void* const* d_in, const int* in_sizes, int n_in,
                              void* d_out, int out_size, void* d_ws, size_t ws_size,
                              hipStream_t stream) {
  (void)in_sizes; (void)n_in; (void)out_size; (void)ws_size;

  float* ws = (float*)d_ws;
  float* X    = ws;                    // grid activations  (max 393216)
  float* XN   = X    + 393216;
  float* XW   = XN   + 393216;         // windows (max 889056)
  float* XW2  = XW   + 1048576;        // AV (proj input)
  float* QKV  = XW2  + 1048576;        // (max 2667168)
  float* ATT  = QKV  + 2700000;        // (max 19059138)
  float* MLPH = ATT  + 19200000;       // (max 1572864)
  float* M8   = MLPH + 1600000;        // merged 8C (max 393216)
  float* HD   = M8   + 400000;
  float* feat = HD;            // 2x768
  float* fcl  = feat + 1536;
  float* t0   = fcl  + 1536;
  float* t1   = t0   + 1536;
  float* zb   = t1   + 1536;   // 2x1024
  float* h1   = zb   + 2048;   // 2x512

  patch_embed_kernel<<<dim3(512, 3), 32, 0, stream>>>(PW(0), PW(5), PW(4), X);

  int D = 16;
  for (int l = 0; l < 4; ++l) {
    int c = 48 << l, heads = 3 << l;
    int wsz = (l < 2) ? 7 : (l == 2 ? 4 : 2);
    int ssb = (l < 2) ? 3 : 0;
    int pd = ((D + wsz - 1) / wsz) * wsz;
    int nwd = pd / wsz, nw = nwd * nwd * nwd, n = wsz * wsz * wsz;
    int tokens = 2 * D * D * D;
    int nwt = 2 * nw;
    int mtiles = (n + 15) / 16;

    for (int blk = 0; blk < 2; ++blk) {
      int pb = 18 + (l * 2 + blk) * 13;
      const float* fc1_b = PW(pb + 0); const float* fc1_w = PW(pb + 1);
      const float* fc2_b = PW(pb + 2); const float* fc2_w = PW(pb + 3);
      const float* n1_b  = PW(pb + 4); const float* n1_g  = PW(pb + 5);
      const float* n2_b  = PW(pb + 6); const float* n2_g  = PW(pb + 7);
      const float* pj_b  = PW(pb + 8); const float* pj_w  = PW(pb + 9);
      const float* qk_b  = PW(pb + 10); const float* qk_w = PW(pb + 11);
      const float* rpb   = PW(pb + 12);
      int ss = (blk & 1) ? ssb : 0;
      int shifted = (ss > 0) ? 1 : 0;

      layernorm_kernel<<<tokens, 32, 0, stream>>>(X, n1_g, n1_b, XN, c);
      win_gather_kernel<<<dim3(nwt, n), 32, 0, stream>>>(XN, XW, D, c, wsz, ss, pd);
      {
        int M = nwt * n, N = 3 * c, K = c;
        gemm_xwt_kernel<0><<<dim3((M + 15) / 16, (N + 15) / 16), 32, 0, stream>>>(
            XW, qk_w, qk_b, QKV, M, N, K, N);
      }
      attn_score_kernel<<<dim3(mtiles, mtiles, nwt * heads), 32, 0, stream>>>(
          QKV, rpb, ATT, n, c, heads, nwd, wsz, ss, pd, shifted);
      softmax_kernel<<<nwt * heads * n, 32, 0, stream>>>(ATT, n);
      attn_av_kernel<<<dim3(mtiles, nwt * heads), 32, 0, stream>>>(ATT, QKV, XW2, n, c, heads);
      {
        int M = nwt * n;
        gemm_xwt_kernel<0><<<dim3((M + 15) / 16, (c + 15) / 16), 32, 0, stream>>>(
            XW2, pj_w, pj_b, XW, M, c, c, c);
      }
      win_scatter_kernel<<<tokens, 32, 0, stream>>>(XW, X, D, c, wsz, ss, pd);

      layernorm_kernel<<<tokens, 32, 0, stream>>>(X, n2_g, n2_b, XN, c);
      gemm_xwt_kernel<1><<<dim3((tokens + 15) / 16, (4 * c + 15) / 16), 32, 0, stream>>>(
          XN, fc1_w, fc1_b, MLPH, tokens, 4 * c, c, 4 * c);
      gemm_xwt_kernel<0><<<dim3((tokens + 15) / 16, (c + 15) / 16), 32, 0, stream>>>(
          MLPH, fc2_w, fc2_b, XW, tokens, c, 4 * c, c);
      add_kernel<<<((size_t)tokens * c + 255) / 256, 256, 0, stream>>>(X, XW, (size_t)tokens * c);
    }

    int mb = 122 + l * 3;
    const float* mn_b = PW(mb + 0); const float* mn_g = PW(mb + 1); const float* red_w = PW(mb + 2);
    int Dh = D / 2, t2 = 2 * Dh * Dh * Dh;
    merge_gather_kernel<<<t2, 32, 0, stream>>>(X, M8, D, c);
    layernorm_kernel<<<t2, 32, 0, stream>>>(M8, mn_g, mn_b, XW, 8 * c);
    gemm_xwt_kernel<0><<<dim3((t2 + 15) / 16, (2 * c + 15) / 16), 32, 0, stream>>>(
        XW, red_w, nullptr, X, t2, 2 * c, 8 * c, 2 * c);
    D = Dh;
  }

  // ---- head: X is (2, 768) ----
  layernorm_kernel<<<2, 32, 0, stream>>>(X, nullptr, nullptr, feat, 768);
  gemm_xwt_kernel<0><<<dim3(1, 48), 32, 0, stream>>>(PW(1), PW(3), PW(2), fcl, 2, 768, 2, 768);
  gemm_xwt_kernel<0><<<dim3(1, 48), 32, 0, stream>>>(feat, PW(11), PW(10), t0, 2, 768, 768, 768);
  gemm_xwt_kernel<0><<<dim3(1, 32), 32, 0, stream>>>(t0, PW(13), PW(12), zb, 2, 512, 768, 1024);
  gemm_xwt_kernel<0><<<dim3(1, 48), 32, 0, stream>>>(fcl, PW(15), PW(14), t1, 2, 768, 768, 768);
  gemm_xwt_kernel<0><<<dim3(1, 32), 32, 0, stream>>>(t1, PW(17), PW(16), zb + 512, 2, 512, 768, 1024);
  gemm_xwt_kernel<0><<<dim3(1, 32), 32, 0, stream>>>(zb, PW(7), PW(6), h1, 2, 512, 1024, 512);
  gemm_xwt_kernel<0><<<dim3(1, 1), 32, 0, stream>>>(h1, PW(9), PW(8), (float*)d_out, 2, 1, 512, 1);
}
